// BiMambaHead_54589034332652
// MI455X (gfx1250) — matrix operand order repaired and checked by reading, not compiled, not run on hardware
//
#include <hip/hip_runtime.h>

// ---------------------------------------------------------------------------
// BiMamba2 head for MI455X (gfx1250, wave32, WMMA).
// GEMMs: bf16 WMMA 16x16x32, f32 accumulation, double-buffered LDS tiles
// staged via async global->LDS copies (ASYNCcnt) when the toolchain exposes
// the gfx1250 builtin, else synchronous b128 staging.
// ---------------------------------------------------------------------------

typedef __bf16 bf16x8  __attribute__((ext_vector_type(8)));
typedef __bf16 bf16x16 __attribute__((ext_vector_type(16)));
typedef float  f32x8   __attribute__((ext_vector_type(8)));

#define AS1 __attribute__((address_space(1)))
#define AS3 __attribute__((address_space(3)))

// Parameter type of the gfx1250 async-LDS builtins (per hipcc diagnostic):
// pointer to 16-byte GCC vector of int, AS1 (global) / AS3 (LDS).
typedef int v4i_b128 __attribute__((__vector_size__(4 * sizeof(int))));

#if defined(__AMDGCN__) && defined(__gfx1250__) && \
    __has_builtin(__builtin_amdgcn_global_load_async_to_lds_b128) && \
    __has_builtin(__builtin_amdgcn_s_wait_asynccnt)
#define USE_ASYNC_LDS 1
#else
#define USE_ASYNC_LDS 0
#endif

#define MTOK   32768      // B*L
#define LSEQ   1024
#define NB     32
#define DMODEL 512
#define DINNER 1024
#define DSTATE 16
#define NHEADS 16
#define CONVD  1056       // DINNER + 2*DSTATE
#define DPROJ  2096       // 2*DINNER + 2*DSTATE + NHEADS

#define BM 128
#define BN 64
#define BK 32
#define LDK 40            // LDS row stride in bf16 (16B aligned, deconflicted)

__device__ __forceinline__ float silu_f(float x) {
    return x * (1.0f / (1.0f + __expf(-x)));
}

// ------------------------------ f32 -> bf16 --------------------------------
__global__ void k_f32_to_bf16(const float* __restrict__ in,
                              __bf16* __restrict__ out, int n) {
    int i = blockIdx.x * blockDim.x + threadIdx.x;
    if (i < n) out[i] = (__bf16)in[i];
}

// ------------------------------ WMMA GEMM ----------------------------------
// C[M,N] = A[M,K](bf16) * W[N,K](bf16)^T  (+ bias) (+ old C if accumulate)
// grid: (ceil(N/BN), M/BM), block: 256 = 8 waves.
// Wave w computes rows [w*16, w*16+16) of the 128-row block tile, all 4
// 16-wide N sub-tiles. Double-buffered LDS, async-staged tiles.
__global__ __launch_bounds__(256) void k_gemm_bf16(
    const __bf16* __restrict__ A, const __bf16* __restrict__ W,
    float* __restrict__ C, const float* __restrict__ bias,
    int M, int N, int K, int accumulate)
{
    __shared__ __bf16 As[2][BM * LDK];
    __shared__ __bf16 Bs[2][BN * LDK];

    const int tid  = threadIdx.x;
    const int wave = tid >> 5;
    const int lane = tid & 31;
    const int m0   = blockIdx.y * BM;
    const int n0   = blockIdx.x * BN;

    // Per-thread staging coordinates.
    const int br   = tid >> 2;            // B tile row (0..63)
    const int bc8  = (tid & 3) << 3;      // B tile col chunk (bf16 units)
    const bool bvalid = (n0 + br) < N;

    // Out-of-range B rows stay zero for the whole K loop (staged rows only
    // ever overwrite valid rows).
    if (!bvalid) {
        #pragma unroll
        for (int j = 0; j < 8; ++j) {
            Bs[0][br * LDK + bc8 + j] = (__bf16)0.0f;
            Bs[1][br * LDK + bc8 + j] = (__bf16)0.0f;
        }
    }

    f32x8 acc[4];
    #pragma unroll
    for (int t = 0; t < 4; ++t)
        #pragma unroll
        for (int j = 0; j < 8; ++j) acc[t][j] = 0.0f;

    auto stage = [&](int buf, int k0) {
#if USE_ASYNC_LDS
        #pragma unroll
        for (int i = 0; i < 2; ++i) {
            int chunk = tid + i * 256;
            int r  = chunk >> 2;
            int c8 = (chunk & 3) << 3;
            __builtin_amdgcn_global_load_async_to_lds_b128(
                (AS1 v4i_b128*)(void*)&A[(size_t)(m0 + r) * K + k0 + c8],
                (AS3 v4i_b128*)(void*)&As[buf][r * LDK + c8], 0, 0);
        }
        if (bvalid) {
            __builtin_amdgcn_global_load_async_to_lds_b128(
                (AS1 v4i_b128*)(void*)&W[(size_t)(n0 + br) * K + k0 + bc8],
                (AS3 v4i_b128*)(void*)&Bs[buf][br * LDK + bc8], 0, 0);
        }
#else
        #pragma unroll
        for (int i = 0; i < 2; ++i) {
            int chunk = tid + i * 256;
            int r  = chunk >> 2;
            int c8 = (chunk & 3) << 3;
            bf16x8 v = *(const bf16x8*)&A[(size_t)(m0 + r) * K + k0 + c8];
            *(bf16x8*)&As[buf][r * LDK + c8] = v;
        }
        if (bvalid) {
            bf16x8 v = *(const bf16x8*)&W[(size_t)(n0 + br) * K + k0 + bc8];
            *(bf16x8*)&Bs[buf][br * LDK + bc8] = v;
        }
#endif
    };

    const int nk = K / BK;
    stage(0, 0);
#if USE_ASYNC_LDS
    __builtin_amdgcn_s_wait_asynccnt(0);
#endif
    __syncthreads();

    for (int kt = 0; kt < nk; ++kt) {
        const int cur = kt & 1;
        if (kt + 1 < nk) stage((kt + 1) & 1, (kt + 1) * BK);
        if (kt + 2 < nk) {
            __builtin_prefetch(&A[(size_t)(m0 + (tid >> 1)) * K + (kt + 2) * BK], 0, 1);
        }

        // A fragment (16x32 bf16, wave32 layout):
        //   lanes 0-15 : row m, K = 0..7 (v0..3), 16..23 (v4..7)
        //   lanes 16-31: row m, K = 8..15,        24..31
        const int mrow = (wave << 4) + (lane & 15);
        const int ka   = (lane < 16) ? 0 : 8;
        bf16x8 alo = *(const bf16x8*)&As[cur][mrow * LDK + ka];
        bf16x8 ahi = *(const bf16x8*)&As[cur][mrow * LDK + ka + 16];
        bf16x16 afrag = __builtin_shufflevector(
            alo, ahi, 0,1,2,3,4,5,6,7,8,9,10,11,12,13,14,15);

        // B fragment (32x16 bf16): lanes 0-15 -> K=0..7,16..23, lanes 16-31 -> K=8..15,24..31
        const int kb = (lane < 16) ? 0 : 8;
        #pragma unroll
        for (int nt = 0; nt < 4; ++nt) {
            const int nrow = (nt << 4) + (lane & 15);
            bf16x8 blo = *(const bf16x8*)&Bs[cur][nrow * LDK + kb];
            bf16x8 bhi = *(const bf16x8*)&Bs[cur][nrow * LDK + kb + 16];
            bf16x16 bfrag = __builtin_shufflevector(
                blo, bhi, 0,1,2,3,4,5,6,7,8,9,10,11,12,13,14,15);
            acc[nt] = __builtin_amdgcn_wmma_f32_16x16x32_bf16(
                false, afrag, false, bfrag, (short)0, acc[nt], false, false);
        }

#if USE_ASYNC_LDS
        if (kt + 1 < nk) __builtin_amdgcn_s_wait_asynccnt(0);
#endif
        __syncthreads();
    }

    // Store: C/D layout -> VGPR i: lanes0-15 M=i, lanes16-31 M=8+i; N = lane&15.
    const int mbase = m0 + (wave << 4) + ((lane >> 4) << 3);
    const int ncol  = lane & 15;
    for (int nt = 0; nt < 4; ++nt) {
        int col = n0 + (nt << 4) + ncol;
        if (col < N) {
            float bv = bias ? bias[col] : 0.0f;
            #pragma unroll
            for (int i = 0; i < 8; ++i) {
                size_t idx = (size_t)(mbase + i) * N + col;
                float v = acc[nt][i] + bv;
                if (accumulate) v += C[idx];
                C[idx] = v;
            }
        }
    }
}

// --------------------------- dt / dA precompute ----------------------------
__global__ void k_dt(const float* __restrict__ zx,
                     const float* __restrict__ dt_bias,
                     const float* __restrict__ A_log,
                     float* __restrict__ dt, float* __restrict__ dA) {
    int i = blockIdx.x * blockDim.x + threadIdx.x;
    if (i >= MTOK * NHEADS) return;
    int h = i & (NHEADS - 1);
    int row = i >> 4;
    float x = zx[(size_t)row * DPROJ + (DINNER + CONVD) + h] + dt_bias[h];
    float sp = (x > 20.0f) ? x : log1pf(__expf(x));
    float A = -__expf(A_log[h]);
    dt[i] = sp;
    dA[i] = __expf(sp * A);
}

// ------------------------- depthwise causal conv ---------------------------
// fwd: v[t] = silu(b + sum_j w[j]*u[t-3+j]);  bwd: v[t] = silu(b + sum_j w[j]*u[t+3-j])
__global__ void k_conv(const float* __restrict__ zx,
                       const float* __restrict__ cw,   // [1056,4]
                       const float* __restrict__ cb,   // [1056]
                       float* __restrict__ out,        // [MTOK,1056]
                       int reverse) {
    int idx = blockIdx.x * blockDim.x + threadIdx.x;
    if (idx >= MTOK * CONVD) return;
    int c  = idx % CONVD;
    int bl = idx / CONVD;
    int t  = bl & (LSEQ - 1);
    int b  = bl >> 10;
    float acc = cb[c];
    #pragma unroll
    for (int j = 0; j < 4; ++j) {
        int tt = reverse ? (t + 3 - j) : (t - 3 + j);
        if (tt >= 0 && tt < LSEQ) {
            acc += cw[c * 4 + j] *
                   zx[((size_t)(b << 10) + tt) * DPROJ + DINNER + c];
        }
    }
    out[idx] = silu_f(acc);
}

// ------------------------------- SSM scan ----------------------------------
// One block per (b,h). 256 threads: p = tid>>2 (0..63), n-group = tid&3 (4 states).
// State lives fully in registers; 4-wide C-reduction via wave32 shfl_xor.
__global__ __launch_bounds__(256) void k_scan(
    const float* __restrict__ conv,   // [MTOK, 1056]
    const float* __restrict__ dt,     // [MTOK, 16]
    const float* __restrict__ dA,     // [MTOK, 16]
    const float* __restrict__ Dv,     // [16]
    float* __restrict__ y,            // [MTOK, 1024]
    int reverse)
{
    int bh = blockIdx.x;
    int b = bh >> 4;
    int h = bh & 15;
    int tid = threadIdx.x;
    int p = tid >> 2;
    int ng = tid & 3;
    float hs0 = 0.f, hs1 = 0.f, hs2 = 0.f, hs3 = 0.f;
    float Dh = Dv[h];

    for (int s = 0; s < LSEQ; ++s) {
        int t = reverse ? (LSEQ - 1 - s) : s;
        size_t row = (size_t)b * LSEQ + t;
        const float* cr = conv + row * CONVD;
        float xv = cr[(h << 6) + p];
        float4 Bv = *(const float4*)(cr + DINNER + (ng << 2));
        float4 Cv = *(const float4*)(cr + DINNER + DSTATE + (ng << 2));
        float dtv = dt[row * NHEADS + h];
        float dAv = dA[row * NHEADS + h];
        float cx = dtv * xv;
        hs0 = hs0 * dAv + cx * Bv.x;
        hs1 = hs1 * dAv + cx * Bv.y;
        hs2 = hs2 * dAv + cx * Bv.z;
        hs3 = hs3 * dAv + cx * Bv.w;
        float yp = hs0 * Cv.x + hs1 * Cv.y + hs2 * Cv.z + hs3 * Cv.w;
        yp += __shfl_xor(yp, 1, 32);
        yp += __shfl_xor(yp, 2, 32);
        if (ng == 0) y[row * DINNER + (h << 6) + p] = yp + Dh * xv;
    }
}

// ------------------- gate (silu(z)) + RMS norm -> bf16 ---------------------
__global__ __launch_bounds__(256) void k_gatenorm(
    const float* __restrict__ y, const float* __restrict__ zx,
    const float* __restrict__ norm_w, __bf16* __restrict__ out) {
    __shared__ float red[8];
    size_t row = blockIdx.x;
    int tid = threadIdx.x;
    float v[4];
    float ss = 0.0f;
    #pragma unroll
    for (int i = 0; i < 4; ++i) {
        int c = tid + i * 256;
        float z = zx[row * DPROJ + c];
        float t = y[row * DINNER + c] * silu_f(z);
        v[i] = t;
        ss += t * t;
    }
    #pragma unroll
    for (int o = 16; o > 0; o >>= 1) ss += __shfl_xor(ss, o, 32);
    if ((tid & 31) == 0) red[tid >> 5] = ss;
    __syncthreads();
    if (tid == 0) {
        float tot = 0.0f;
        for (int i = 0; i < 8; ++i) tot += red[i];
        red[0] = rsqrtf(tot * (1.0f / (float)DINNER) + 1e-5f);
    }
    __syncthreads();
    float scale = red[0];
    #pragma unroll
    for (int i = 0; i < 4; ++i) {
        int c = tid + i * 256;
        out[row * DINNER + c] = (__bf16)(v[i] * scale * norm_w[c]);
    }
}

// ---------------------------------------------------------------------------
extern "C" void kernel_launch(void* const* d_in, const int* in_sizes, int n_in,
                              void* d_out, int out_size, void* d_ws, size_t ws_size,
                              hipStream_t stream) {
    const float* x          = (const float*)d_in[0];
    const float* in_proj_w  = (const float*)d_in[1];
    const float* conv_w     = (const float*)d_in[2];
    const float* conv_b     = (const float*)d_in[3];
    const float* dt_bias    = (const float*)d_in[4];
    const float* A_log      = (const float*)d_in[5];
    const float* Dv         = (const float*)d_in[6];
    const float* norm_w     = (const float*)d_in[7];
    const float* out_proj_w = (const float*)d_in[8];
    const float* fc_w       = (const float*)d_in[9];
    const float* fc_b       = (const float*)d_in[10];
    float* out = (float*)d_out;

    char* ws = (char*)d_ws;
    size_t off = 0;
    auto alloc = [&](size_t bytes) -> void* {
        void* p = ws + off;
        off += (bytes + 255) & ~(size_t)255;
        return p;
    };

    __bf16* xbf   = (__bf16*)alloc((size_t)MTOK * DMODEL * 2);
    __bf16* winb  = (__bf16*)alloc((size_t)DPROJ * DMODEL * 2);
    __bf16* woutb = (__bf16*)alloc((size_t)DMODEL * DINNER * 2);
    __bf16* wfcb  = (__bf16*)alloc((size_t)96 * DMODEL * 2);
    float*  zx    = (float*)alloc((size_t)MTOK * DPROJ * 4);
    float*  dtb   = (float*)alloc((size_t)MTOK * NHEADS * 4);
    float*  dab   = (float*)alloc((size_t)MTOK * NHEADS * 4);
    float*  convb = (float*)alloc((size_t)MTOK * CONVD * 4);
    float*  yb    = (float*)alloc((size_t)MTOK * DINNER * 4);
    __bf16* ynb   = (__bf16*)alloc((size_t)MTOK * DINNER * 2);
    float*  osum  = (float*)alloc((size_t)MTOK * DMODEL * 4);
    __bf16* osb   = (__bf16*)alloc((size_t)MTOK * DMODEL * 2);

    const int TB = 256;
    // Convert activations and weights to bf16 once.
    k_f32_to_bf16<<<(MTOK * DMODEL + TB - 1) / TB, TB, 0, stream>>>(x, xbf, MTOK * DMODEL);
    k_f32_to_bf16<<<(DPROJ * DMODEL + TB - 1) / TB, TB, 0, stream>>>(in_proj_w, winb, DPROJ * DMODEL);
    k_f32_to_bf16<<<(DMODEL * DINNER + TB - 1) / TB, TB, 0, stream>>>(out_proj_w, woutb, DMODEL * DINNER);
    k_f32_to_bf16<<<(96 * DMODEL + TB - 1) / TB, TB, 0, stream>>>(fc_w, wfcb, 96 * DMODEL);

    // GEMM1: zxbcdt = x @ in_proj_w^T   (shared by both directions)
    {
        dim3 g((DPROJ + BN - 1) / BN, MTOK / BM);
        k_gemm_bf16<<<g, TB, 0, stream>>>(xbf, winb, zx, nullptr,
                                          MTOK, DPROJ, DMODEL, 0);
    }

    k_dt<<<(MTOK * NHEADS + TB - 1) / TB, TB, 0, stream>>>(zx, dt_bias, A_log, dtb, dab);

    for (int dir = 0; dir < 2; ++dir) {
        k_conv<<<(MTOK * CONVD + TB - 1) / TB, TB, 0, stream>>>(
            zx, conv_w, conv_b, convb, dir);
        k_scan<<<NB * NHEADS, TB, 0, stream>>>(convb, dtb, dab, Dv, yb, dir);
        k_gatenorm<<<MTOK, TB, 0, stream>>>(yb, zx, norm_w, ynb);
        // GEMM2: osum (+)= ynorm @ out_proj_w^T
        dim3 g2(DMODEL / BN, MTOK / BM);
        k_gemm_bf16<<<g2, TB, 0, stream>>>(ynb, woutb, osum, nullptr,
                                           MTOK, DMODEL, DINNER, dir);
    }

    k_f32_to_bf16<<<(MTOK * DMODEL + TB - 1) / TB, TB, 0, stream>>>(osum, osb, MTOK * DMODEL);

    // GEMM3: out = (out_f + out_b) @ fc_w^T + fc_b
    {
        dim3 g3((96 + BN - 1) / BN, MTOK / BM);
        k_gemm_bf16<<<g3, TB, 0, stream>>>(osb, wfcb, out, fc_b,
                                           MTOK, 96, DMODEL, 0);
    }
}
